// BiLSTM_CRF_1924145348913
// MI455X (gfx1250) — matrix operand order repaired and checked by reading, not compiled, not run on hardware
//
#include <hip/hip_runtime.h>
#include <hip/hip_fp16.h>

typedef __attribute__((ext_vector_type(16))) _Float16 v16h;
typedef __attribute__((ext_vector_type(8)))  float    v8f;
typedef __attribute__((ext_vector_type(4)))  _Float16 h4;

#define NB   256
#define NT   512
#define NK   128
#define START_TAG 126
#define STOP_TAG  127
#define NEGV (-10000.0f)

// ---------------------------------------------------------------------------
// Forward algorithm as a WMMA scan:
//   alpha_new[to,b] = m_b + log( sum_from ET[to,from] * p[from,b] ) + feat
//   ET = exp(trans) (fixed, f16)   p = exp(alpha - m) (f16, in [0,1])
// Each block: 16 batch columns; 8 waves, each owning a 16(to)x16(b) tile.
// A fragments (ET) live in VGPRs for all 512 steps. alpha/p are stored
// TRANSPOSED in LDS so B-fragment loads are contiguous 32B (2x ds_load_b128).
// ---------------------------------------------------------------------------
__global__ __launch_bounds__(256)
void crf_forward_wmma(const float* __restrict__ feats,
                      const float* __restrict__ trans,
                      float* __restrict__ logZ)
{
    __shared__ _Float16 ET[NK][136];                              // exp(trans)
    __shared__ __attribute__((aligned(32))) float    alphaT[16][136]; // [b][from]
    __shared__ __attribute__((aligned(32))) _Float16 pT[16][144];     // [b][from]
    __shared__ __attribute__((aligned(32))) float    featl[16][132];  // [b][to]
    __shared__ float mpart[16][17];
    __shared__ float mcol[16];

    const int tid = threadIdx.x;
    const int bg0 = blockIdx.x * 16;

    // build ET = exp(trans) in LDS (NEG rows -> 0)
    for (int i = tid; i < NK * NK; i += 256)
        ET[i >> 7][i & 127] = (_Float16)__expf(trans[i]);
    // init alphaT
    for (int i = tid; i < 16 * NK; i += 256) {
        const int b = i >> 7, fr = i & 127;
        alphaT[b][fr] = (fr == START_TAG) ? 0.0f : NEGV;
    }
    __syncthreads();

    const int lane = tid & 31;
    const int wv   = tid >> 5;      // wave id 0..7 -> to tile
    const int half = lane >> 4;
    const int l15  = lane & 15;
    const int arow = wv * 16 + l15; // A-matrix row (ISA 16-bit A layout)

    // Preload loop-invariant A fragments (ET rows for this wave's to-tile)
    v16h afrag[4];
#pragma unroll
    for (int c4 = 0; c4 < 4; ++c4) {
        const int kk = c4 * 32;
#pragma unroll
        for (int j = 0; j < 16; ++j) {
            const int k = kk + half * 16 + j;
            afrag[c4][j] = ET[arow][k];
        }
    }

    // per-thread contiguous slice for the m / p phases
    const int pb = tid >> 4;            // batch col 0..15
    const int f0 = (tid & 15) * 8;      // 8 contiguous 'from'

    for (int t = 0; t < NT; ++t) {
        // phase A: load feature tile (coalesced) + partial row max of alphaT
        for (int i = tid; i < 2048; i += 256) {
            const int bl = i >> 7, to = i & 127;
            featl[bl][to] = feats[((size_t)(bg0 + bl) * NT + t) * NK + to];
            if (t + 1 < NT)
                __builtin_prefetch(&feats[((size_t)(bg0 + bl) * NT + t + 1) * NK + to], 0, 1);
        }
        {
            const float4 x0 = *(const float4*)&alphaT[pb][f0];
            const float4 x1 = *(const float4*)&alphaT[pb][f0 + 4];
            float mx = fmaxf(fmaxf(fmaxf(x0.x, x0.y), fmaxf(x0.z, x0.w)),
                             fmaxf(fmaxf(x1.x, x1.y), fmaxf(x1.z, x1.w)));
            mpart[tid & 15][pb] = mx;
        }
        __syncthreads();
        if (tid < 16) {
            float mx = mpart[0][tid];
#pragma unroll
            for (int s = 1; s < 16; ++s) mx = fmaxf(mx, mpart[s][tid]);
            mcol[tid] = mx;
        }
        __syncthreads();
        // phase B: pT = exp(alphaT - m), 8 contiguous halves per thread
        {
            const float mb = mcol[pb];
            const float4 x0 = *(const float4*)&alphaT[pb][f0];
            const float4 x1 = *(const float4*)&alphaT[pb][f0 + 4];
            h4 y0, y1;
            y0[0] = (_Float16)__expf(x0.x - mb); y0[1] = (_Float16)__expf(x0.y - mb);
            y0[2] = (_Float16)__expf(x0.z - mb); y0[3] = (_Float16)__expf(x0.w - mb);
            y1[0] = (_Float16)__expf(x1.x - mb); y1[1] = (_Float16)__expf(x1.y - mb);
            y1[2] = (_Float16)__expf(x1.z - mb); y1[3] = (_Float16)__expf(x1.w - mb);
            *(h4*)&pT[pb][f0]     = y0;
            *(h4*)&pT[pb][f0 + 4] = y1;
        }
        __syncthreads();

        // phase C: D = ET_tile x P   (4 chunks of K=32)
        // B fragment = pT[l15][kk + half*16 .. +15]  -> one aligned 32B read
        v8f c = {};
#pragma unroll
        for (int c4 = 0; c4 < 4; ++c4) {
            const int kk = c4 * 32;
            const v16h bf = *(const v16h*)&pT[l15][kk + half * 16];
            c = __builtin_amdgcn_wmma_f32_16x16x32_f16(
                    false, afrag[c4], false, bf, (short)0, c, false, false);
        }
        // writeback: alpha_new = m + log(s) + feat  (C/D layout: row=half*8+r)
        {
            const float mb = mcol[l15];
            const int to0 = wv * 16 + half * 8;
            const float4 fa = *(const float4*)&featl[l15][to0];
            const float4 fb = *(const float4*)&featl[l15][to0 + 4];
            float4 r0, r1;
            r0.x = mb + __logf(c[0]) + fa.x;  r0.y = mb + __logf(c[1]) + fa.y;
            r0.z = mb + __logf(c[2]) + fa.z;  r0.w = mb + __logf(c[3]) + fa.w;
            r1.x = mb + __logf(c[4]) + fb.x;  r1.y = mb + __logf(c[5]) + fb.y;
            r1.z = mb + __logf(c[6]) + fb.z;  r1.w = mb + __logf(c[7]) + fb.w;
            *(float4*)&alphaT[l15][to0]     = r0;
            *(float4*)&alphaT[l15][to0 + 4] = r1;
        }
        __syncthreads();
    }

    // logZ[b] = LSE_to( alphaT[b][to] + trans[STOP,to] )
    if (tid < 16) {
        const int b = tid;
        float mx = -3.4e38f;
        for (int to = 0; to < NK; ++to)
            mx = fmaxf(mx, alphaT[b][to] + trans[STOP_TAG * NK + to]);
        float s = 0.f;
        for (int to = 0; to < NK; ++to)
            s += __expf(alphaT[b][to] + trans[STOP_TAG * NK + to] - mx);
        logZ[bg0 + b] = mx + __logf(s);
    }
}

// ---------------------------------------------------------------------------
// Viterbi: max-plus scan (no tensor-core semiring) -> vectorized VALU.
// vT is transposed so each thread streams contiguous float4; trans rows are
// read as 8-byte half vectors. 1024 threads: b=tid&15, 2 'to' rows/thread.
// ---------------------------------------------------------------------------
__global__ __launch_bounds__(1024)
void crf_viterbi(const float* __restrict__ feats,
                 const float* __restrict__ trans,
                 unsigned char* __restrict__ bptr,
                 float* __restrict__ path_score,   // d_out + 1
                 int* __restrict__ lastTag)
{
    __shared__ __attribute__((aligned(16))) _Float16 th[NK][136];
    __shared__ __attribute__((aligned(16))) float    vT[16][136];  // [b][from]
    __shared__ __attribute__((aligned(16))) float    featl[16][132];

    const int tid = threadIdx.x;
    const int bg0 = blockIdx.x * 16;
    const int b   = tid & 15;
    const int g   = tid >> 4;     // 0..63
    const int to0 = g * 2;

    for (int i = tid; i < NK * NK; i += 1024)
        th[i >> 7][i & 127] = (_Float16)trans[i];
    for (int i = tid; i < 16 * NK; i += 1024)
        vT[i >> 7][i & 127] = ((i & 127) == START_TAG) ? 0.0f : NEGV;
    __syncthreads();

    const _Float16* t0p = &th[to0][0];
    const _Float16* t1p = &th[to0 + 1][0];

    for (int t = 0; t < NT; ++t) {
        for (int i = tid; i < 2048; i += 1024)
            featl[i >> 7][i & 127] =
                feats[((size_t)(bg0 + (i >> 7)) * NT + t) * NK + (i & 127)];
        __syncthreads();

        float m0 = -3.4e38f, m1 = -3.4e38f;
        int   a0 = 0, a1 = 0;
        for (int fr = 0; fr < NK; fr += 4) {
            const float4 va = *(const float4*)&vT[b][fr];
            const h4 h0 = *(const h4*)&t0p[fr];
            const h4 h1 = *(const h4*)&t1p[fr];
            const float vaA[4] = { va.x, va.y, va.z, va.w };
#pragma unroll
            for (int j = 0; j < 4; ++j) {
                const float s0 = vaA[j] + (float)h0[j];
                const float s1 = vaA[j] + (float)h1[j];
                if (s0 > m0) { m0 = s0; a0 = fr + j; }
                if (s1 > m1) { m1 = s1; a1 = fr + j; }
            }
        }
        __syncthreads();                       // all reads of vT done
        vT[b][to0]     = m0 + featl[b][to0];
        vT[b][to0 + 1] = m1 + featl[b][to0 + 1];
        const unsigned short pk = (unsigned short)((a0 & 0xff) | ((a1 & 0xff) << 8));
        *(unsigned short*)(bptr + ((size_t)t * NB + (bg0 + b)) * NK + to0) = pk;
        __syncthreads();                       // writes visible before next iter
    }

    if (tid < 16) {
        float mx = -3.4e38f; int arg = 0;
        for (int to = 0; to < NK; ++to) {
            const float s = vT[tid][to] + trans[STOP_TAG * NK + to];
            if (s > mx) { mx = s; arg = to; }
        }
        path_score[bg0 + tid] = mx;
        lastTag[bg0 + tid]    = arg;
    }
}

// ---------------------------------------------------------------------------
// Tail: gold-path score, backtrack, loss reduction.
// ---------------------------------------------------------------------------
__global__ __launch_bounds__(256)
void crf_tail(const float* __restrict__ feats, const int* __restrict__ tags,
              const float* __restrict__ trans, const float* __restrict__ logZ,
              const int* __restrict__ lastTag,
              const unsigned char* __restrict__ bptr,
              float* __restrict__ out)
{
    __shared__ float red[256];
    const int b = threadIdx.x;

    // gold path score
    float gold = 0.f; int prev = START_TAG;
    for (int t = 0; t < NT; ++t) {
        const int tg = tags[b * NT + t];
        gold += trans[tg * NK + prev] + feats[((size_t)b * NT + t) * NK + tg];
        prev = tg;
    }
    gold += trans[STOP_TAG * NK + prev];
    red[b] = logZ[b] - gold;

    // backtrack: pred[t] = carry before consuming bptr[t]
    int cur = lastTag[b];
    for (int t = NT - 1; t >= 0; --t) {
        out[257 + (size_t)b * NT + t] = (float)cur;
        cur = bptr[((size_t)t * NB + b) * NK + cur];
    }
    __syncthreads();

    for (int s = 128; s > 0; s >>= 1) {
        if (b < s) red[b] += red[b + s];
        __syncthreads();
    }
    if (b == 0) out[0] = red[0] / (float)(NB * NT);
}

extern "C" void kernel_launch(void* const* d_in, const int* in_sizes, int n_in,
                              void* d_out, int out_size, void* d_ws, size_t ws_size,
                              hipStream_t stream) {
    (void)in_sizes; (void)n_in; (void)out_size; (void)ws_size;
    const float* feats = (const float*)d_in[0];
    const int*   tags  = (const int*)d_in[1];
    const float* trans = (const float*)d_in[2];
    float* out = (float*)d_out;

    char* ws = (char*)d_ws;
    unsigned char* bptr = (unsigned char*)ws;                       // 16 MB
    float* logZ    = (float*)(ws + (size_t)NB * NT * NK);           // 1 KB
    int*   lastTag = (int*)  (ws + (size_t)NB * NT * NK + 1024);    // 1 KB

    crf_forward_wmma<<<NB / 16, 256, 0, stream>>>(feats, trans, logZ);
    crf_viterbi<<<NB / 16, 1024, 0, stream>>>(feats, trans, bptr, out + 1, lastTag);
    crf_tail<<<1, 256, 0, stream>>>(feats, tags, trans, logZ, lastTag, bptr, out);
}
